// MoE_Layer_new_43069932044308
// MI455X (gfx1250) — matrix operand order repaired and checked by reading, not compiled, not run on hardware
//
#include <hip/hip_runtime.h>
#include <hip/hip_bf16.h>

// ---------------------------------------------------------------------------
// MoE layer (dense, 8 experts) for MI455X / gfx1250 (wave32, WMMA bf16).
//   gate = softmax(x @ Wg + bg)              [N,8]
//   h_e  = relu(x @ W1[e] + b1[e])           [N,2048]  (bf16 staging)
//   out  = sum_e gate[:,e] * (h_e @ W2[e] + b2[e])     [N,1024] f32
// GEMMs: v_wmma_f32_16x16x32_bf16, double-buffered LDS fed by
// global_load_async_to_lds_b128 (ASYNCcnt pipeline).
// ---------------------------------------------------------------------------

typedef __bf16        v16bf  __attribute__((ext_vector_type(16)));
typedef float         v8f    __attribute__((ext_vector_type(8)));
typedef unsigned int  u32x4  __attribute__((ext_vector_type(4)));

__device__ __forceinline__ unsigned short f2bf(float f) {
    unsigned int u = __float_as_uint(f);
    unsigned int r = u + 0x7FFFu + ((u >> 16) & 1u);   // round-to-nearest-even
    return (unsigned short)(r >> 16);
}

__device__ __forceinline__ v16bf load_frag(const unsigned char* p0,
                                           const unsigned char* p1) {
    union { u32x4 q[2]; v16bf v; } u;
    u.q[0] = *(const u32x4*)p0;   // ds_load_b128
    u.q[1] = *(const u32x4*)p1;   // ds_load_b128
    return u.v;
}

// CDNA5 async global->LDS copy, 16B per lane. vdst = LDS byte address.
__device__ __forceinline__ void async_copy16(unsigned lds_off, const void* g) {
    asm volatile("global_load_async_to_lds_b128 %0, %1, off"
                 :: "v"(lds_off), "v"(g) : "memory");
}
__device__ __forceinline__ void wait_async_le8() {
    asm volatile("s_wait_asynccnt 0x8" ::: "memory");
}
__device__ __forceinline__ void wait_async_0() {
    asm volatile("s_wait_asynccnt 0x0" ::: "memory");
}

// ---------------------------------------------------------------------------
// Transpose + fp32->bf16 convert:  in [E][R][C] f32  ->  out [E][C][R] bf16
// ---------------------------------------------------------------------------
__global__ __launch_bounds__(256) void transcvt_kernel(
    const float* __restrict__ in, unsigned short* __restrict__ out,
    int R, int C)
{
    __shared__ float tile[32][33];
    const int e  = blockIdx.z;
    const int c0 = blockIdx.x * 32;
    const int r0 = blockIdx.y * 32;
    const float*    src = in  + (size_t)e * R * C;
    unsigned short* dst = out + (size_t)e * R * C;
    const int tx = threadIdx.x & 31;
    const int ty = threadIdx.x >> 5;
#pragma unroll
    for (int i = 0; i < 32; i += 8)
        tile[ty + i][tx] = src[(size_t)(r0 + ty + i) * C + (c0 + tx)];
    __syncthreads();
#pragma unroll
    for (int i = 0; i < 32; i += 8)
        dst[(size_t)(c0 + ty + i) * R + (r0 + tx)] = f2bf(tile[tx][ty + i]);
}

// ---------------------------------------------------------------------------
// Gating: one wave32 per token. softmax(x@Wg+bg), plus bf16 copy of x.
// ---------------------------------------------------------------------------
__global__ __launch_bounds__(256) void gate_kernel(
    const float* __restrict__ x, const float* __restrict__ Wg,
    const float* __restrict__ bg, unsigned short* __restrict__ xbf,
    float* __restrict__ gate, int Ntok, int Dd)
{
    const int wv   = (int)((blockIdx.x * blockDim.x + threadIdx.x) >> 5);
    const int lane = (int)(threadIdx.x & 31);
    if (wv >= Ntok) return;
    const float*    xr = x   + (size_t)wv * Dd;
    unsigned short* xo = xbf + (size_t)wv * Dd;

    float acc[8] = {0.f,0.f,0.f,0.f,0.f,0.f,0.f,0.f};
    for (int d = lane; d < Dd; d += 32) {
        float xv = xr[d];
        xo[d] = f2bf(xv);
        const float* wr = Wg + (size_t)d * 8;
#pragma unroll
        for (int e = 0; e < 8; ++e) acc[e] = fmaf(xv, wr[e], acc[e]);
    }
#pragma unroll
    for (int e = 0; e < 8; ++e) {
        float v = acc[e];
#pragma unroll
        for (int off = 16; off > 0; off >>= 1) v += __shfl_xor(v, off, 32);
        acc[e] = v;
    }
    if (lane == 0) {
        float mx = -3.4e38f;
#pragma unroll
        for (int e = 0; e < 8; ++e) { acc[e] += bg[e]; mx = fmaxf(mx, acc[e]); }
        float s = 0.f;
#pragma unroll
        for (int e = 0; e < 8; ++e) { acc[e] = __expf(acc[e] - mx); s += acc[e]; }
        float inv = 1.f / s;
#pragma unroll
        for (int e = 0; e < 8; ++e) gate[(size_t)wv * 8 + e] = acc[e] * inv;
    }
}

// ---------------------------------------------------------------------------
// GEMM geometry: block tile 128x128, K-stage 64, 256 threads = 8 waves (4x2),
// wave tile 32x64 = 2x4 WMMA tiles, 16 WMMA per stage per wave.
// LDS: 2 stages x (A 128x144B + B 128x144B) = 72 KB, async-filled.
// ---------------------------------------------------------------------------
#define LSTRIDE 144                    // 64 bf16 = 128B + 16B pad (16B aligned)
#define TILE_BYTES (128 * LSTRIDE)     // 18432
#define STAGE_BYTES (2 * TILE_BYTES)   // 36864

// GEMM1: Hout[M,N] = relu(A[M,K] @ Bt[N,K]^T + bias[N]); bf16 in/out, f32 acc
__global__ __launch_bounds__(256) void gemm1_kernel(
    const unsigned short* __restrict__ A,     // xbf   [M, K]   K = 1024
    const unsigned short* __restrict__ Bt,    // W1^T  [N, K]   (N = H)
    const float* __restrict__ bias,           // b1[e] [N]
    unsigned short* __restrict__ Hout,        // [M, N] bf16
    int Nn, int Kd)
{
    __shared__ __align__(16) unsigned char lds[2 * STAGE_BYTES];
    const unsigned lds_base = (unsigned)(size_t)(void*)&lds[0];

    const int bm   = blockIdx.y * 128;
    const int bn   = blockIdx.x * 128;
    const int tid  = threadIdx.x;
    const int lane = tid & 31;
    const int wave = tid >> 5;
    const int wm   = (wave >> 1) * 32;
    const int wn   = (wave & 1) * 64;
    const int lh   = lane & 15;
    const int kh   = lane >> 4;

    const unsigned char* Ab = (const unsigned char*)A;
    const unsigned char* Bb = (const unsigned char*)Bt;

    auto issue = [&](int k0, int buf) {
        const unsigned baseA = lds_base + buf * STAGE_BYTES;
        const unsigned baseB = baseA + TILE_BYTES;
#pragma unroll
        for (int c = 0; c < 4; ++c) {
            int chunk = tid + c * 256;           // 1024 chunks of 16B per tile
            int row   = chunk >> 3;
            int off   = (chunk & 7) * 16;
            async_copy16(baseA + row * LSTRIDE + off,
                         Ab + ((size_t)(bm + row) * Kd + k0) * 2 + off);
            async_copy16(baseB + row * LSTRIDE + off,
                         Bb + ((size_t)(bn + row) * Kd + k0) * 2 + off);
        }
    };

    v8f acc[2][4];
#pragma unroll
    for (int r = 0; r < 2; ++r)
#pragma unroll
        for (int t = 0; t < 4; ++t) acc[r][t] = (v8f)0.f;

    const int nk = Kd >> 6;                      // stages of K=64
    issue(0, 0);
    if (nk > 1) issue(64, 1);

    for (int k = 0; k < nk; ++k) {
        if (k + 1 < nk) wait_async_le8();        // stage k landed, k+1 in flight
        else            wait_async_0();
        __syncthreads();

        const unsigned char* sA = lds + (k & 1) * STAGE_BYTES;
        const unsigned char* sB = sA + TILE_BYTES;
#pragma unroll
        for (int ks = 0; ks < 2; ++ks) {         // two K=32 WMMA steps
            v16bf af[2], bf[4];
#pragma unroll
            for (int r = 0; r < 2; ++r) {
                const unsigned char* p =
                    sA + (wm + r * 16 + lh) * LSTRIDE + ks * 64 + kh * 16;
                af[r] = load_frag(p, p + 32);
            }
#pragma unroll
            for (int t = 0; t < 4; ++t) {
                const unsigned char* p =
                    sB + (wn + t * 16 + lh) * LSTRIDE + ks * 64 + kh * 16;
                bf[t] = load_frag(p, p + 32);
            }
#pragma unroll
            for (int r = 0; r < 2; ++r)
#pragma unroll
                for (int t = 0; t < 4; ++t)
                    acc[r][t] = __builtin_amdgcn_wmma_f32_16x16x32_bf16(
                        false, af[r], false, bf[t], (short)0, acc[r][t],
                        false, false);
        }
        __syncthreads();
        if (k + 2 < nk) issue((k + 2) << 6, k & 1);
    }

    // epilogue: +bias, relu, bf16 store
#pragma unroll
    for (int t = 0; t < 4; ++t) {
        const int   n  = bn + wn + t * 16 + lh;
        const float bv = bias[n];
#pragma unroll
        for (int r = 0; r < 2; ++r) {
#pragma unroll
            for (int v = 0; v < 8; ++v) {
                int   m   = bm + wm + r * 16 + kh * 8 + v;
                float val = acc[r][t][v] + bv;
                val = val > 0.f ? val : 0.f;
                Hout[(size_t)m * Nn + n] = f2bf(val);
            }
        }
    }
}

// GEMM2: out[M,N] (+)= gate[m] * (A[M,K] @ Bt[N,K]^T + bias[N]); f32 out
__global__ __launch_bounds__(256) void gemm2_kernel(
    const unsigned short* __restrict__ A,     // hbuf  [M, K]  K = 2048
    const unsigned short* __restrict__ Bt,    // W2^T  [N, K]  (N = D)
    const float* __restrict__ bias,           // b2[e] [N]
    const float* __restrict__ gate_e,         // &gate[0][e], stride 8
    float* __restrict__ Out,                  // [M, N] f32
    int Nn, int Kd, int first)
{
    __shared__ __align__(16) unsigned char lds[2 * STAGE_BYTES];
    __shared__ float sg[128];
    __shared__ float sb[128];
    const unsigned lds_base = (unsigned)(size_t)(void*)&lds[0];

    const int bm   = blockIdx.y * 128;
    const int bn   = blockIdx.x * 128;
    const int tid  = threadIdx.x;
    const int lane = tid & 31;
    const int wave = tid >> 5;
    const int wm   = (wave >> 1) * 32;
    const int wn   = (wave & 1) * 64;
    const int lh   = lane & 15;
    const int kh   = lane >> 4;

    if (tid < 128)       sg[tid]       = gate_e[(size_t)(bm + tid) * 8];
    else                 sb[tid - 128] = bias[bn + tid - 128];

    const unsigned char* Ab = (const unsigned char*)A;
    const unsigned char* Bb = (const unsigned char*)Bt;

    auto issue = [&](int k0, int buf) {
        const unsigned baseA = lds_base + buf * STAGE_BYTES;
        const unsigned baseB = baseA + TILE_BYTES;
#pragma unroll
        for (int c = 0; c < 4; ++c) {
            int chunk = tid + c * 256;
            int row   = chunk >> 3;
            int off   = (chunk & 7) * 16;
            async_copy16(baseA + row * LSTRIDE + off,
                         Ab + ((size_t)(bm + row) * Kd + k0) * 2 + off);
            async_copy16(baseB + row * LSTRIDE + off,
                         Bb + ((size_t)(bn + row) * Kd + k0) * 2 + off);
        }
    };

    v8f acc[2][4];
#pragma unroll
    for (int r = 0; r < 2; ++r)
#pragma unroll
        for (int t = 0; t < 4; ++t) acc[r][t] = (v8f)0.f;

    const int nk = Kd >> 6;
    issue(0, 0);
    if (nk > 1) issue(64, 1);

    for (int k = 0; k < nk; ++k) {
        if (k + 1 < nk) wait_async_le8();
        else            wait_async_0();
        __syncthreads();

        const unsigned char* sA = lds + (k & 1) * STAGE_BYTES;
        const unsigned char* sB = sA + TILE_BYTES;
#pragma unroll
        for (int ks = 0; ks < 2; ++ks) {
            v16bf af[2], bf[4];
#pragma unroll
            for (int r = 0; r < 2; ++r) {
                const unsigned char* p =
                    sA + (wm + r * 16 + lh) * LSTRIDE + ks * 64 + kh * 16;
                af[r] = load_frag(p, p + 32);
            }
#pragma unroll
            for (int t = 0; t < 4; ++t) {
                const unsigned char* p =
                    sB + (wn + t * 16 + lh) * LSTRIDE + ks * 64 + kh * 16;
                bf[t] = load_frag(p, p + 32);
            }
#pragma unroll
            for (int r = 0; r < 2; ++r)
#pragma unroll
                for (int t = 0; t < 4; ++t)
                    acc[r][t] = __builtin_amdgcn_wmma_f32_16x16x32_bf16(
                        false, af[r], false, bf[t], (short)0, acc[r][t],
                        false, false);
        }
        __syncthreads();
        if (k + 2 < nk) issue((k + 2) << 6, k & 1);
    }

    // epilogue: gate[m] * (acc + b2[n]); write (expert 0) or accumulate
#pragma unroll
    for (int t = 0; t < 4; ++t) {
        const int   n  = bn + wn + t * 16 + lh;
        const float bv = sb[wn + t * 16 + lh];
#pragma unroll
        for (int r = 0; r < 2; ++r) {
#pragma unroll
            for (int v = 0; v < 8; ++v) {
                int    mrow = wm + r * 16 + kh * 8 + v;
                float  g    = sg[mrow];
                float  val  = g * (acc[r][t][v] + bv);
                size_t idx  = (size_t)(bm + mrow) * Nn + n;
                if (first) Out[idx] = val;
                else       Out[idx] = Out[idx] + val;
            }
        }
    }
}

// ---------------------------------------------------------------------------
// Launcher
// ---------------------------------------------------------------------------
extern "C" void kernel_launch(void* const* d_in, const int* in_sizes, int n_in,
                              void* d_out, int out_size, void* d_ws, size_t ws_size,
                              hipStream_t stream) {
    constexpr int Bb = 4, T = 2048, Dd = 1024, Hh = 2048, Ee = 8;
    constexpr int Ntok = Bb * T;                       // 8192

    const float* x  = (const float*)d_in[0];
    const float* Wg = (const float*)d_in[1];
    const float* bg = (const float*)d_in[2];
    const float* W1 = (const float*)d_in[3];
    const float* b1 = (const float*)d_in[4];
    const float* W2 = (const float*)d_in[5];
    const float* b2 = (const float*)d_in[6];
    float* out = (float*)d_out;

    unsigned char* ws = (unsigned char*)d_ws;
    unsigned short* xbf  = (unsigned short*)ws;  ws += (size_t)Ntok * Dd * 2;       // 16 MB
    unsigned short* w1t  = (unsigned short*)ws;  ws += (size_t)Ee * Hh * Dd * 2;    // 32 MB
    unsigned short* w2t  = (unsigned short*)ws;  ws += (size_t)Ee * Dd * Hh * 2;    // 32 MB
    unsigned short* hbuf = (unsigned short*)ws;  ws += (size_t)Ntok * Hh * 2;       // 32 MB
    float*          gate = (float*)ws;                                              // 256 KB

    // W1 [E][D][H] -> w1t [E][H][D];  W2 [E][H][D] -> w2t [E][D][H]
    transcvt_kernel<<<dim3(Hh / 32, Dd / 32, Ee), 256, 0, stream>>>(W1, w1t, Dd, Hh);
    transcvt_kernel<<<dim3(Dd / 32, Hh / 32, Ee), 256, 0, stream>>>(W2, w2t, Hh, Dd);

    gate_kernel<<<Ntok / 8, 256, 0, stream>>>(x, Wg, bg, xbf, gate, Ntok, Dd);

    for (int e = 0; e < Ee; ++e) {
        gemm1_kernel<<<dim3(Hh / 128, Ntok / 128), 256, 0, stream>>>(
            xbf, w1t + (size_t)e * Hh * Dd, b1 + (size_t)e * Hh, hbuf, Hh, Dd);
        gemm2_kernel<<<dim3(Dd / 128, Ntok / 128), 256, 0, stream>>>(
            hbuf, w2t + (size_t)e * Dd * Hh, b2 + (size_t)e * Dd, gate + e, out,
            Dd, Hh, e == 0 ? 1 : 0);
    }
}
